// WrapEvoformerLayer_1417339207826
// MI455X (gfx1250) — compile-verified, matrix operand order checked by reading, not run on hardware
//
#include <hip/hip_runtime.h>
#include <math.h>

// ---------------------------------------------------------------------------
// Types / WMMA helpers (CDNA5 gfx1250, wave32)
// ---------------------------------------------------------------------------
typedef __attribute__((ext_vector_type(16))) __bf16 bf16x16;
typedef __attribute__((ext_vector_type(8)))  float  f32x8;
typedef __attribute__((ext_vector_type(4)))  unsigned int u32x4;

union Frag16 { bf16x16 v; u32x4 u[2]; __bf16 e[16]; };

__device__ __forceinline__ f32x8 wmma_bf16(const Frag16& a, const Frag16& b, f32x8 c) {
  return __builtin_amdgcn_wmma_f32_16x16x32_bf16(false, a.v, false, b.v, (short)0, c,
                                                 false, false);
}
__device__ __forceinline__ float sigf(float x) { return 1.f / (1.f + __expf(-x)); }

// Async global -> LDS copy (CDNA5 TDM-adjacent path, tracked by ASYNCcnt).
// One instruction moves 32 lanes x 16B. LDS address = low 32 bits of the
// generic pointer to __shared__ memory.
__device__ __forceinline__ void async_ld_b128(const void* gaddr, void* lds) {
  unsigned l = (unsigned)(uintptr_t)lds;
  asm volatile("global_load_async_to_lds_b128 %0, %1, off"
               :: "v"(l), "v"(gaddr) : "memory");
}
__device__ __forceinline__ void wait_async0() {
  asm volatile("s_wait_asynccnt 0x0" ::: "memory");
}
__device__ __forceinline__ void wait_async1() {
  asm volatile("s_wait_asynccnt 0x1" ::: "memory");
}

enum { EPI_NONE = 0, EPI_RELU = 1, EPI_SIG = 2 };

// ---------------------------------------------------------------------------
// Generic GEMM: C[M,N] = act(A[M,K] * Bt[N,K]^T + bias) (+ residual)
// A, Bt bf16 (Bt N-major). Block = 256 threads = 8 waves; wave tile 16x32;
// block tile 128(M)x32(N). The 32xK B panel is shared by all 8 waves, so it
// is staged through LDS with double-buffered async loads (1 async instr per
// wave per 64-wide K tile); A fragments stream from global per wave.
// M % 128 == 0, N % 32 == 0, K % 64 == 0 at all call sites.
// ---------------------------------------------------------------------------
__global__ __launch_bounds__(256) void gemm_bf16_kernel(
    const __bf16* __restrict__ A, const __bf16* __restrict__ Bt,
    const float* __restrict__ bias, const float* __restrict__ residual,
    float* __restrict__ outF, __bf16* __restrict__ outH,
    int M, int N, int K, int epi) {
  __shared__ __bf16 sB[2][32 * 64];
  const int lane = threadIdx.x & 31, wave = threadIdx.x >> 5;
  const int lm = lane & 15, lh = lane >> 4;
  const int m0 = blockIdx.y * 128 + wave * 16;
  const int n0 = blockIdx.x * 32;
  const __bf16* aRow = A + (size_t)(m0 + lm) * K;
  // staging geometry: wave w stages panel rows [4w, 4w+4); lane -> (row, 16B col)
  const int srow = 4 * wave + (lane >> 3);
  const int scol = (lane & 7) * 8;
  const __bf16* gB = Bt + (size_t)(n0 + srow) * K + scol;
  f32x8 acc0 = {0.f,0.f,0.f,0.f,0.f,0.f,0.f,0.f};
  f32x8 acc1 = acc0;
  const int NK = K >> 6;
  async_ld_b128(gB, &sB[0][srow * 64 + scol]);
  for (int kt = 0; kt < NK; ++kt) {
    if (kt + 1 < NK) {
      async_ld_b128(gB + (kt + 1) * 64, &sB[(kt + 1) & 1][srow * 64 + scol]);
      __builtin_prefetch(aRow + (kt + 1) * 64, 0, 0);
      wait_async1();
    } else {
      wait_async0();
    }
    __syncthreads();
    const __bf16* bb = sB[kt & 1];
    const __bf16* aK = aRow + kt * 64;
#pragma unroll
    for (int s2 = 0; s2 < 2; ++s2) {
      Frag16 a, b0, b1;
      a.u[0]  = *(const u32x4*)(aK + s2 * 32 + 8 * lh);
      a.u[1]  = *(const u32x4*)(aK + s2 * 32 + 16 + 8 * lh);
      b0.u[0] = *(const u32x4*)(bb + lm * 64        + s2 * 32 + 8 * lh);
      b0.u[1] = *(const u32x4*)(bb + lm * 64        + s2 * 32 + 16 + 8 * lh);
      b1.u[0] = *(const u32x4*)(bb + (16 + lm) * 64 + s2 * 32 + 8 * lh);
      b1.u[1] = *(const u32x4*)(bb + (16 + lm) * 64 + s2 * 32 + 16 + 8 * lh);
      acc0 = wmma_bf16(a, b0, acc0);
      acc1 = wmma_bf16(a, b1, acc1);
    }
    __syncthreads();
  }
  for (int t = 0; t < 2; ++t) {
    f32x8 acc = t ? acc1 : acc0;
    int n = n0 + t * 16 + lm;
    float bv = bias ? bias[n] : 0.f;
#pragma unroll
    for (int r = 0; r < 8; ++r) {
      int m = m0 + r + 8 * lh;
      size_t idx = (size_t)m * N + n;
      float v = acc[r] + bv;
      if (epi == EPI_RELU)      v = v > 0.f ? v : 0.f;
      else if (epi == EPI_SIG)  v = sigf(v);
      if (residual) v += residual[idx];
      if (outF) outF[idx] = v;
      else      outH[idx] = (__bf16)v;
    }
  }
}

// ---------------------------------------------------------------------------
// Fused self-attention: one wave per (batch-row b, head h, 16-query tile).
// Q,K bf16 [NB,L,H*32]; Vt bf16 [NB,H,32,L]; bias1 [NB,L]; bias2 [H,L,L]|null
// logits(WMMA) -> LDS -> softmax (lane-pair reduction) -> AV(WMMA) -> O bf16
// ---------------------------------------------------------------------------
__global__ __launch_bounds__(32) void attn_kernel(
    const __bf16* __restrict__ Q, const __bf16* __restrict__ Kmat,
    const __bf16* __restrict__ Vt, const float* __restrict__ bias1,
    const float* __restrict__ bias2, __bf16* __restrict__ O,
    int L, int H, float scale) {
  __shared__ float sm[16 * 256];
  const int lane = threadIdx.x, lm = lane & 15, lh = lane >> 4;
  const int qt = blockIdx.x, h = blockIdx.y, b = blockIdx.z;
  const int HC = H * 32;
  const __bf16* qRow = Q + ((size_t)b * L + qt * 16 + lm) * HC + h * 32;
  Frag16 aq;
  aq.u[0] = *(const u32x4*)(qRow + 8 * lh);
  aq.u[1] = *(const u32x4*)(qRow + 16 + 8 * lh);
  const float* b1 = bias1 + (size_t)b * L;
  for (int kt = 0; kt < L / 16; ++kt) {
    const __bf16* kRow = Kmat + ((size_t)b * L + kt * 16 + lm) * HC + h * 32;
    Frag16 bk;
    bk.u[0] = *(const u32x4*)(kRow + 8 * lh);
    bk.u[1] = *(const u32x4*)(kRow + 16 + 8 * lh);
    f32x8 acc = {0.f,0.f,0.f,0.f,0.f,0.f,0.f,0.f};
    acc = wmma_bf16(aq, bk, acc);
    int n = kt * 16 + lm;
    float bn = b1[n];
    const float* b2 = bias2 ? bias2 + ((size_t)h * L + qt * 16) * L + n : nullptr;
#pragma unroll
    for (int r = 0; r < 8; ++r) {
      int m = r + 8 * lh;
      float v = acc[r] * scale + bn;
      if (b2) v += b2[(size_t)m * L];
      sm[m * L + n] = v;
    }
  }
  __syncthreads();
  {  // softmax: 16 rows; lanes l and l+16 own the two halves of row l%16
    int hl = L / 2;
    int base = lm * L + lh * hl;
    float mx = -3.0e38f;
    for (int j = 0; j < hl; ++j) mx = fmaxf(mx, sm[base + j]);
    mx = fmaxf(mx, __shfl_xor(mx, 16, 32));
    float s = 0.f;
    for (int j = 0; j < hl; ++j) { float e = __expf(sm[base + j] - mx); sm[base + j] = e; s += e; }
    s += __shfl_xor(s, 16, 32);
    float inv = 1.f / s;
    for (int j = 0; j < hl; ++j) sm[base + j] *= inv;
  }
  __syncthreads();
  f32x8 o0 = {0.f,0.f,0.f,0.f,0.f,0.f,0.f,0.f};
  f32x8 o1 = o0;
  const __bf16* vBase = Vt + ((size_t)b * H + h) * 32 * L;
  for (int kt = 0; kt < L / 32; ++kt) {
    Frag16 ap;
#pragma unroll
    for (int e = 0; e < 16; ++e) {
      int vv = e >> 1;
      int koff = (vv < 4 ? 2 * vv : 16 + 2 * (vv - 4)) + 8 * lh + (e & 1);
      ap.e[e] = (__bf16)sm[lm * L + kt * 32 + koff];
    }
    const __bf16* v0 = vBase + (size_t)lm * L + kt * 32;
    const __bf16* v1 = vBase + (size_t)(16 + lm) * L + kt * 32;
    Frag16 bv0, bv1;
    bv0.u[0] = *(const u32x4*)(v0 + 8 * lh); bv0.u[1] = *(const u32x4*)(v0 + 16 + 8 * lh);
    bv1.u[0] = *(const u32x4*)(v1 + 8 * lh); bv1.u[1] = *(const u32x4*)(v1 + 16 + 8 * lh);
    o0 = wmma_bf16(ap, bv0, o0);
    o1 = wmma_bf16(ap, bv1, o1);
  }
#pragma unroll
  for (int r = 0; r < 8; ++r) {
    int q = qt * 16 + r + 8 * lh;
    size_t orow = ((size_t)b * L + q) * HC + h * 32;
    O[orow + lm]      = (__bf16)o0[r];
    O[orow + 16 + lm] = (__bf16)o1[r];
  }
}

// ---------------------------------------------------------------------------
// Triangle-mul einsum: per channel c, out[i,j,c] = sum_k At[c,i,k]*Bt[c,j,k]
// 128 batched 256x256x256 GEMMs (z = channel). Same async-staged B panel.
// ---------------------------------------------------------------------------
__global__ __launch_bounds__(256) void trimul_gemm_kernel(
    const __bf16* __restrict__ At, const __bf16* __restrict__ Bt,
    float* __restrict__ out, int R) {
  __shared__ __bf16 sB[2][32 * 64];
  const int c = blockIdx.z;
  const __bf16* A = At + (size_t)c * R * R;
  const __bf16* B = Bt + (size_t)c * R * R;
  const int lane = threadIdx.x & 31, wave = threadIdx.x >> 5;
  const int lm = lane & 15, lh = lane >> 4;
  const int m0 = blockIdx.y * 128 + wave * 16;
  const int n0 = blockIdx.x * 32;
  const __bf16* aRow = A + (size_t)(m0 + lm) * R;
  const int srow = 4 * wave + (lane >> 3);
  const int scol = (lane & 7) * 8;
  const __bf16* gB = B + (size_t)(n0 + srow) * R + scol;
  f32x8 acc0 = {0.f,0.f,0.f,0.f,0.f,0.f,0.f,0.f};
  f32x8 acc1 = acc0;
  const int NK = R >> 6;
  async_ld_b128(gB, &sB[0][srow * 64 + scol]);
  for (int kt = 0; kt < NK; ++kt) {
    if (kt + 1 < NK) {
      async_ld_b128(gB + (kt + 1) * 64, &sB[(kt + 1) & 1][srow * 64 + scol]);
      __builtin_prefetch(aRow + (kt + 1) * 64, 0, 0);
      wait_async1();
    } else {
      wait_async0();
    }
    __syncthreads();
    const __bf16* bb = sB[kt & 1];
    const __bf16* aK = aRow + kt * 64;
#pragma unroll
    for (int s2 = 0; s2 < 2; ++s2) {
      Frag16 a, b0, b1;
      a.u[0]  = *(const u32x4*)(aK + s2 * 32 + 8 * lh);
      a.u[1]  = *(const u32x4*)(aK + s2 * 32 + 16 + 8 * lh);
      b0.u[0] = *(const u32x4*)(bb + lm * 64        + s2 * 32 + 8 * lh);
      b0.u[1] = *(const u32x4*)(bb + lm * 64        + s2 * 32 + 16 + 8 * lh);
      b1.u[0] = *(const u32x4*)(bb + (16 + lm) * 64 + s2 * 32 + 8 * lh);
      b1.u[1] = *(const u32x4*)(bb + (16 + lm) * 64 + s2 * 32 + 16 + 8 * lh);
      acc0 = wmma_bf16(a, b0, acc0);
      acc1 = wmma_bf16(a, b1, acc1);
    }
    __syncthreads();
  }
  for (int t = 0; t < 2; ++t) {
    f32x8 acc = t ? acc1 : acc0;
    int n = n0 + t * 16 + lm;
#pragma unroll
    for (int r = 0; r < 8; ++r) {
      int m = m0 + r + 8 * lh;
      out[((size_t)m * R + n) * 128 + c] = acc[r];
    }
  }
}

// ---------------------------------------------------------------------------
// Outer-product-mean: one block (4 waves) per (i,j).
// Both 32x128 panels are shared by all 4 waves -> async-stage to LDS, then
// outer[32,32] = at[i]·bt[j]^T via WMMA, then 1024->128 projection + norm.
// at/bt bf16 [R,32,S], S == 128.
// ---------------------------------------------------------------------------
__global__ __launch_bounds__(128) void opm_kernel(
    const __bf16* __restrict__ At, const __bf16* __restrict__ Btm,
    const float* __restrict__ ow, const float* __restrict__ ob,
    const float* __restrict__ nmask, float* __restrict__ pair, int R, int S) {
  __shared__ __bf16 sA[32 * 128];
  __shared__ __bf16 sBp[32 * 128];
  __shared__ float souter[1024];
  __shared__ float rnorm[128];
  const int i = blockIdx.x, j = blockIdx.y;
  const int lane = threadIdx.x & 31, wave = threadIdx.x >> 5;
  const int lm = lane & 15, lh = lane >> 4;
  const int mt = wave >> 1, nt = wave & 1;
  const __bf16* gA = At  + (size_t)i * 32 * S;
  const __bf16* gB = Btm + (size_t)j * 32 * S;
  // 32 chunks of 512B (16 per panel); wave w stages chunks w, w+4, ...
  for (int t = 0; t < 8; ++t) {
    int chunk = wave + 4 * t;
    int isB = chunk >> 4;
    int c2 = chunk & 15;
    int row = c2 * 2 + (lane >> 4);
    int col = (lane & 15) * 8;
    const __bf16* g = (isB ? gB : gA) + (size_t)row * S + col;
    __bf16* l = (isB ? sBp : sA) + row * 128 + col;
    async_ld_b128(g, l);
  }
  wait_async0();
  __syncthreads();
  f32x8 acc = {0.f,0.f,0.f,0.f,0.f,0.f,0.f,0.f};
  const __bf16* aBase = sA  + (mt * 16 + lm) * 128;
  const __bf16* bBase = sBp + (nt * 16 + lm) * 128;
#pragma unroll
  for (int k0 = 0; k0 < 128; k0 += 32) {
    Frag16 a, b;
    a.u[0] = *(const u32x4*)(aBase + k0 + 8 * lh);
    a.u[1] = *(const u32x4*)(aBase + k0 + 16 + 8 * lh);
    b.u[0] = *(const u32x4*)(bBase + k0 + 8 * lh);
    b.u[1] = *(const u32x4*)(bBase + k0 + 16 + 8 * lh);
    acc = wmma_bf16(a, b, acc);
  }
#pragma unroll
  for (int r = 0; r < 8; ++r) {
    int c = mt * 16 + r + 8 * lh;
    int d = nt * 16 + lm;
    souter[c * 32 + d] = acc[r];
  }
  rnorm[threadIdx.x] = (threadIdx.x < (unsigned)S)
      ? nmask[(size_t)threadIdx.x * R + i] * nmask[(size_t)threadIdx.x * R + j] : 0.f;
  __syncthreads();
  for (int off = 64; off > 0; off >>= 1) {
    if ((int)threadIdx.x < off) rnorm[threadIdx.x] += rnorm[threadIdx.x + off];
    __syncthreads();
  }
  float norm = rnorm[0];
  int p = threadIdx.x;
  float sum = 0.f;
  for (int cd = 0; cd < 1024; ++cd) sum += souter[cd] * ow[(size_t)cd * 128 + p];
  pair[((size_t)i * R + j) * 128 + p] += sum / (norm + 1e-3f) + ob[p];
}

// ---------------------------------------------------------------------------
// Small helper kernels (LN, transposes, elementwise, weight conversion)
// ---------------------------------------------------------------------------
__global__ void ln_kernel(const float* __restrict__ x, const float* __restrict__ g,
                          const float* __restrict__ b, __bf16* __restrict__ y, int D) {
  size_t row = blockIdx.x;
  const float* xr = x + row * D;
  float s = 0.f, s2 = 0.f;
  for (int i = threadIdx.x; i < D; i += 128) { float v = xr[i]; s += v; s2 += v * v; }
  __shared__ float rs[128], rq[128];
  rs[threadIdx.x] = s; rq[threadIdx.x] = s2;
  __syncthreads();
  for (int off = 64; off > 0; off >>= 1) {
    if ((int)threadIdx.x < off) { rs[threadIdx.x] += rs[threadIdx.x + off]; rq[threadIdx.x] += rq[threadIdx.x + off]; }
    __syncthreads();
  }
  float mean = rs[0] / D;
  float var  = rq[0] / D - mean * mean;
  float inv  = rsqrtf(var + 1e-5f);
  for (int i = threadIdx.x; i < D; i += 128)
    y[row * D + i] = (__bf16)((xr[i] - mean) * inv * g[i] + b[i]);
}

__global__ void convert_wT_kernel(const float* __restrict__ w, __bf16* __restrict__ wt,
                                  int K, int N) {
  int t = blockIdx.x * 256 + threadIdx.x;
  if (t >= K * N) return;
  int n = t / K, k = t % K;
  wt[t] = (__bf16)w[(size_t)k * N + n];
}

__global__ void pb_kernel(const __bf16* __restrict__ X, const float* __restrict__ W,
                          float* __restrict__ out, int Mtot, int K, int H) {
  int t = blockIdx.x * 256 + threadIdx.x;
  if (t >= Mtot * H) return;
  int h = t % H; size_t ij = (size_t)(t / H);
  const __bf16* xr = X + ij * K;
  float s = 0.f;
  for (int k = 0; k < K; ++k) s += (float)xr[k] * W[(size_t)k * H + h];
  out[(size_t)h * Mtot + ij] = s;
}

__global__ void vt_transpose_kernel(const __bf16* __restrict__ v, __bf16* __restrict__ vt,
                                    int NB, int L, int H) {
  size_t t = (size_t)blockIdx.x * 256 + threadIdx.x;
  size_t total = (size_t)NB * L * H * 32;
  if (t >= total) return;
  int c = (int)(t & 31); size_t r = t >> 5;
  int h = (int)(r % H); r /= H;
  int l = (int)(r % L); size_t b = r / L;
  vt[(((size_t)b * H + h) * 32 + c) * L + l] = v[t];
}

__global__ void opm_at_kernel(const float* __restrict__ a, const float* __restrict__ nm,
                              __bf16* __restrict__ at, int S, int R) {
  size_t t = (size_t)blockIdx.x * 256 + threadIdx.x;
  size_t total = (size_t)S * R * 32;
  if (t >= total) return;
  int c = (int)(t & 31); size_t r = t >> 5;
  int rr = (int)(r % R); int s = (int)(r / R);
  at[((size_t)rr * 32 + c) * S + s] = (__bf16)(a[t] * nm[(size_t)s * R + rr]);
}

__global__ void trimul_tr_kernel(const __bf16* __restrict__ in, __bf16* __restrict__ out,
                                 int R, int C, int mode) {
  size_t t = (size_t)blockIdx.x * 256 + threadIdx.x;
  size_t total = (size_t)R * R * C;
  if (t >= total) return;
  int c = (int)(t % C); size_t r = t / C;
  int i1 = (int)(r % R); size_t i0 = r / R;
  size_t oi = (mode == 0) ? (((size_t)c * R + i0) * R + i1)
                          : (((size_t)c * R + i1) * R + i0);
  out[oi] = in[t];
}

__global__ void transpose_pq_kernel(const float* __restrict__ in, float* __restrict__ out,
                                    int P, int Q, int D) {
  size_t t = (size_t)blockIdx.x * 256 + threadIdx.x;
  size_t total = (size_t)P * Q * D;
  if (t >= total) return;
  int d = (int)(t % D); size_t r = t / D;
  int q = (int)(r % Q); size_t p = r / Q;
  out[((size_t)q * P + p) * D + d] = in[t];
}

__global__ void add_transposed_kernel(float* __restrict__ tgt, const float* __restrict__ delta,
                                      int P, int Q, int D) {
  size_t t = (size_t)blockIdx.x * 256 + threadIdx.x;
  size_t total = (size_t)P * Q * D;
  if (t >= total) return;
  int d = (int)(t % D); size_t r = t / D;
  int q = (int)(r % Q); size_t p = r / Q;
  tgt[t] += delta[((size_t)q * P + p) * D + d];
}

__global__ void ew_mul_bb_kernel(const __bf16* __restrict__ a, const __bf16* __restrict__ b,
                                 __bf16* __restrict__ o, size_t n) {
  size_t t = (size_t)blockIdx.x * 256 + threadIdx.x;
  if (t >= n) return;
  o[t] = (__bf16)((float)a[t] * (float)b[t]);
}

__global__ void ew_trimul_kernel(const float* __restrict__ lin, const float* __restrict__ sg,
                                 const float* __restrict__ mask, __bf16* __restrict__ o,
                                 size_t n, int C) {
  size_t t = (size_t)blockIdx.x * 256 + threadIdx.x;
  if (t >= n) return;
  o[t] = (__bf16)(mask[t / C] * sg[t] * lin[t]);
}

__global__ void ew_mul_add_kernel(float* __restrict__ tgt, const float* __restrict__ a,
                                  const float* __restrict__ b, size_t n) {
  size_t t = (size_t)blockIdx.x * 256 + threadIdx.x;
  if (t >= n) return;
  tgt[t] += a[t] * b[t];
}

// ---------------------------------------------------------------------------
// Host-side orchestration
// ---------------------------------------------------------------------------
namespace {

struct Arena { char* base; size_t off, cap; };
static void* aAlloc(Arena& a, size_t bytes) {
  void* p = a.base + a.off;
  a.off += (bytes + 255) & ~(size_t)255;
  return p;
}
static __bf16* aBf(Arena& a, size_t n) { return (__bf16*)aAlloc(a, n * sizeof(__bf16)); }
static float*  aF(Arena& a, size_t n)  { return (float*)aAlloc(a, n * sizeof(float)); }

#define PF(i) ((const float*)d_in[(i)])
#define EWG(n) ((unsigned)(((n) + 255) / 256)), 256

// Param leaf indices (insertion-order recursive flattening of `params`)
enum {
  IN_NODE = 0, IN_PAIR, IN_NODE_MASK, IN_PAIR_MASK, IN_ROW_MASK, IN_COL_MASK,
  IN_TS_MASK, IN_TE_MASK, IN_CHUNK,
  MR_LN_G = 9, MR_LN_B, MR_QW, MR_KW, MR_VW, MR_GW, MR_GB, MR_OW, MR_OB,
  MR_PBLN_G, MR_PBLN_B, MR_PBW,
  MC_LN_G = 21, MC_LN_B, MC_QW, MC_KW, MC_VW, MC_GW, MC_GB, MC_OW, MC_OB,
  MT_BASE = 30,   // ln.g, ln.b, l1.w, l1.b, l2.w, l2.b
  OPM_LN_G = 36, OPM_LN_B, OPM_AW, OPM_AB, OPM_BW, OPM_BB, OPM_OW, OPM_OB,
  TO_BASE = 44,   // trimul: ln_in.g/b, a_p.w/b, a_g.w/b, b_p.w/b, b_g.w/b, ln_out.g/b, o.w/b, g.w/b
  TI_BASE = 60,
  TAS_BASE = 76,  // ln.g/b, q.w, k.w, v.w, g.w/b, o.w/b, pb.w
  TAE_BASE = 86,
  PT_BASE = 96
};

static __bf16* convW(Arena& a, const float* w, int K, int N, hipStream_t s) {
  __bf16* wt = aBf(a, (size_t)K * N);
  int total = K * N;
  convert_wT_kernel<<<(total + 255) / 256, 256, 0, s>>>(w, wt, K, N);
  return wt;
}

static void gemm(hipStream_t s, const __bf16* A, const __bf16* Bt, const float* bias,
                 const float* resid, float* oF, __bf16* oH, size_t M, int N, int K, int epi) {
  dim3 g((unsigned)(N / 32), (unsigned)(M / 128));
  gemm_bf16_kernel<<<g, 256, 0, s>>>(A, Bt, bias, resid, oF, oH, (int)M, N, K, epi);
}

static void lnrun(hipStream_t s, const float* x, const float* g, const float* b,
                  __bf16* y, size_t M, int D) {
  ln_kernel<<<(unsigned)M, 128, 0, s>>>(x, g, b, y, D);
}

// gated self-attention block (LN -> q/k/v/g -> fused attn -> gate -> out-proj)
static void run_attn(Arena ph, hipStream_t s, void* const* d_in,
                     const float* xin, const __bf16* zlPre, int NB, int L, int D, int H,
                     int iLNg, int iLNb, int iQ, int iK, int iV, int iG, int iGb,
                     int iO, int iOb, const float* bias1, const float* bias2,
                     const float* resid, float* outp) {
  size_t M = (size_t)NB * L;
  const __bf16* zl = zlPre;
  if (!zl) {
    __bf16* t = aBf(ph, M * D);
    lnrun(s, xin, PF(iLNg), PF(iLNb), t, M, D);
    zl = t;
  }
  __bf16* Wq = convW(ph, PF(iQ), D, D, s);
  __bf16* Wk = convW(ph, PF(iK), D, D, s);
  __bf16* Wv = convW(ph, PF(iV), D, D, s);
  __bf16* Wg = convW(ph, PF(iG), D, D, s);
  __bf16* q  = aBf(ph, M * D);
  __bf16* k  = aBf(ph, M * D);
  __bf16* v  = aBf(ph, M * D);
  __bf16* gs = aBf(ph, M * D);
  gemm(s, zl, Wq, nullptr, nullptr, nullptr, q, M, D, D, EPI_NONE);
  gemm(s, zl, Wk, nullptr, nullptr, nullptr, k, M, D, D, EPI_NONE);
  gemm(s, zl, Wv, nullptr, nullptr, nullptr, v, M, D, D, EPI_NONE);
  gemm(s, zl, Wg, PF(iGb), nullptr, nullptr, gs, M, D, D, EPI_SIG);
  __bf16* vt = aBf(ph, M * D);
  size_t tot = M * (size_t)D;
  vt_transpose_kernel<<<EWG(tot), 0, s>>>(v, vt, NB, L, H);
  __bf16* o = aBf(ph, M * D);
  attn_kernel<<<dim3((unsigned)(L / 16), (unsigned)H, (unsigned)NB), 32, 0, s>>>(
      q, k, vt, bias1, bias2, o, L, H, 0.17677669529663687f /* 1/sqrt(32) */);
  __bf16* og = aBf(ph, M * D);
  ew_mul_bb_kernel<<<EWG(tot), 0, s>>>(o, gs, og, tot);
  __bf16* Wo = convW(ph, PF(iO), D, D, s);
  gemm(s, og, Wo, PF(iOb), resid, outp, nullptr, M, D, D, EPI_NONE);
}

static void run_transition(Arena ph, hipStream_t s, void* const* d_in,
                           float* buf, size_t M, int D, int NH, int base) {
  __bf16* lnx = aBf(ph, M * D);
  lnrun(s, buf, PF(base + 0), PF(base + 1), lnx, M, D);
  __bf16* W1 = convW(ph, PF(base + 2), D, NH, s);
  __bf16* W2 = convW(ph, PF(base + 4), NH, D, s);
  __bf16* hid = aBf(ph, M * NH);
  gemm(s, lnx, W1, PF(base + 3), nullptr, nullptr, hid, M, NH, D, EPI_RELU);
  gemm(s, hid, W2, PF(base + 5), buf, buf, nullptr, M, D, NH, EPI_NONE);
}

static void run_trimul(Arena ph, hipStream_t s, void* const* d_in,
                       int base, int mode, float* pair, const float* pmask) {
  const int R = 256, D = 128;
  size_t M = (size_t)R * R;
  size_t tot = M * (size_t)D;
  __bf16* zl = aBf(ph, M * D);
  lnrun(s, pair, PF(base + 0), PF(base + 1), zl, M, D);
  float* t1 = aF(ph, M * D);
  float* t2 = aF(ph, M * D);
  __bf16* A  = aBf(ph, M * D);
  __bf16* Bm = aBf(ph, M * D);
  __bf16* Wap = convW(ph, PF(base + 2), D, D, s);
  __bf16* Wag = convW(ph, PF(base + 4), D, D, s);
  gemm(s, zl, Wap, PF(base + 3), nullptr, t1, nullptr, M, D, D, EPI_NONE);
  gemm(s, zl, Wag, PF(base + 5), nullptr, t2, nullptr, M, D, D, EPI_SIG);
  ew_trimul_kernel<<<EWG(tot), 0, s>>>(t1, t2, pmask, A, tot, D);
  __bf16* Wbp = convW(ph, PF(base + 6), D, D, s);
  __bf16* Wbg = convW(ph, PF(base + 8), D, D, s);
  gemm(s, zl, Wbp, PF(base + 7), nullptr, t1, nullptr, M, D, D, EPI_NONE);
  gemm(s, zl, Wbg, PF(base + 9), nullptr, t2, nullptr, M, D, D, EPI_SIG);
  ew_trimul_kernel<<<EWG(tot), 0, s>>>(t1, t2, pmask, Bm, tot, D);
  __bf16* At  = aBf(ph, M * D);
  __bf16* Bt2 = aBf(ph, M * D);
  trimul_tr_kernel<<<EWG(tot), 0, s>>>(A, At, R, D, mode);
  trimul_tr_kernel<<<EWG(tot), 0, s>>>(Bm, Bt2, R, D, mode);
  float* x = aF(ph, M * D);
  trimul_gemm_kernel<<<dim3(R / 32, R / 128, D), 256, 0, s>>>(At, Bt2, x, R);
  __bf16* xln = aBf(ph, M * D);
  lnrun(s, x, PF(base + 10), PF(base + 11), xln, M, D);
  __bf16* Wo = convW(ph, PF(base + 12), D, D, s);
  __bf16* Wg = convW(ph, PF(base + 14), D, D, s);
  gemm(s, xln, Wo, PF(base + 13), nullptr, t1, nullptr, M, D, D, EPI_NONE);
  gemm(s, zl,  Wg, PF(base + 15), nullptr, t2, nullptr, M, D, D, EPI_SIG);
  ew_mul_add_kernel<<<EWG(tot), 0, s>>>(pair, t1, t2, tot);
}

}  // namespace

extern "C" void kernel_launch(void* const* d_in, const int* in_sizes, int n_in,
                              void* d_out, int out_size, void* d_ws, size_t ws_size,
                              hipStream_t stream) {
  (void)in_sizes; (void)n_in; (void)out_size;
  const int S = 128, R = 256, DM = 256, DP = 128;
  const size_t NODE_N = (size_t)S * R * DM;   // 8,388,608
  const size_t PAIR_N = (size_t)R * R * DP;   // 8,388,608

  float* node = (float*)d_out;
  float* pair = node + NODE_N;
  hipMemcpyAsync(node, d_in[IN_NODE], NODE_N * sizeof(float),
                 hipMemcpyDeviceToDevice, stream);
  hipMemcpyAsync(pair, d_in[IN_PAIR], PAIR_N * sizeof(float),
                 hipMemcpyDeviceToDevice, stream);

  Arena ws{(char*)d_ws, 0, ws_size};

  // ---- 1) MSA row attention (pair bias) -> node ----
  {
    Arena ph = ws;
    __bf16* pairln = aBf(ph, PAIR_N);
    lnrun(stream, pair, PF(MR_PBLN_G), PF(MR_PBLN_B), pairln, (size_t)R * R, DP);
    float* zb = aF(ph, (size_t)8 * R * R);
    pb_kernel<<<EWG((size_t)R * R * 8), 0, stream>>>(pairln, PF(MR_PBW), zb, R * R, DP, 8);
    run_attn(ph, stream, d_in, node, nullptr, S, R, DM, 8,
             MR_LN_G, MR_LN_B, MR_QW, MR_KW, MR_VW, MR_GW, MR_GB, MR_OW, MR_OB,
             PF(IN_ROW_MASK), zb, node, node);
  }

  // ---- 2) MSA column attention -> node ----
  {
    Arena ph = ws;
    float* nodeT = aF(ph, NODE_N);
    transpose_pq_kernel<<<EWG(NODE_N), 0, stream>>>(node, nodeT, S, R, DM);
    float* delta = aF(ph, NODE_N);
    run_attn(ph, stream, d_in, nodeT, nullptr, R, S, DM, 8,
             MC_LN_G, MC_LN_B, MC_QW, MC_KW, MC_VW, MC_GW, MC_GB, MC_OW, MC_OB,
             PF(IN_COL_MASK), nullptr, nullptr, delta);
    add_transposed_kernel<<<EWG(NODE_N), 0, stream>>>(node, delta, S, R, DM);
  }

  // ---- 3) MSA transition -> node ----
  run_transition(ws, stream, d_in, node, (size_t)S * R, DM, 4 * DM, MT_BASE);

  // ---- 4) Outer product mean -> pair ----
  {
    Arena ph = ws;
    __bf16* ml = aBf(ph, NODE_N);
    lnrun(stream, node, PF(OPM_LN_G), PF(OPM_LN_B), ml, (size_t)S * R, DM);
    __bf16* Wa = convW(ph, PF(OPM_AW), DM, 32, stream);
    __bf16* Wb = convW(ph, PF(OPM_BW), DM, 32, stream);
    float* af = aF(ph, (size_t)S * R * 32);
    float* bf = aF(ph, (size_t)S * R * 32);
    gemm(stream, ml, Wa, PF(OPM_AB), nullptr, af, nullptr, (size_t)S * R, 32, DM, EPI_NONE);
    gemm(stream, ml, Wb, PF(OPM_BB), nullptr, bf, nullptr, (size_t)S * R, 32, DM, EPI_NONE);
    __bf16* at = aBf(ph, (size_t)R * 32 * S);
    __bf16* bt = aBf(ph, (size_t)R * 32 * S);
    opm_at_kernel<<<EWG((size_t)S * R * 32), 0, stream>>>(af, PF(IN_NODE_MASK), at, S, R);
    opm_at_kernel<<<EWG((size_t)S * R * 32), 0, stream>>>(bf, PF(IN_NODE_MASK), bt, S, R);
    opm_kernel<<<dim3(R, R), 128, 0, stream>>>(at, bt, PF(OPM_OW), PF(OPM_OB),
                                               PF(IN_NODE_MASK), pair, R, S);
  }

  // ---- 5/6) Triangle multiplicative updates -> pair ----
  run_trimul(ws, stream, d_in, TO_BASE, 0, pair, PF(IN_PAIR_MASK));  // outgoing
  run_trimul(ws, stream, d_in, TI_BASE, 1, pair, PF(IN_PAIR_MASK));  // incoming

  // ---- 7) Triangle attention, starting node -> pair ----
  {
    Arena ph = ws;
    __bf16* zl = aBf(ph, PAIR_N);
    lnrun(stream, pair, PF(TAS_BASE + 0), PF(TAS_BASE + 1), zl, (size_t)R * R, DP);
    float* tb = aF(ph, (size_t)4 * R * R);
    pb_kernel<<<EWG((size_t)R * R * 4), 0, stream>>>(zl, PF(TAS_BASE + 9), tb, R * R, DP, 4);
    run_attn(ph, stream, d_in, pair, zl, R, R, DP, 4,
             TAS_BASE + 0, TAS_BASE + 1, TAS_BASE + 2, TAS_BASE + 3, TAS_BASE + 4,
             TAS_BASE + 5, TAS_BASE + 6, TAS_BASE + 7, TAS_BASE + 8,
             PF(IN_TS_MASK), tb, pair, pair);
  }

  // ---- 8) Triangle attention, ending node (transposed) -> pair ----
  {
    Arena ph = ws;
    float* pairT = aF(ph, PAIR_N);
    transpose_pq_kernel<<<EWG(PAIR_N), 0, stream>>>(pair, pairT, R, R, DP);
    __bf16* zl = aBf(ph, PAIR_N);
    lnrun(stream, pairT, PF(TAE_BASE + 0), PF(TAE_BASE + 1), zl, (size_t)R * R, DP);
    float* tb = aF(ph, (size_t)4 * R * R);
    pb_kernel<<<EWG((size_t)R * R * 4), 0, stream>>>(zl, PF(TAE_BASE + 9), tb, R * R, DP, 4);
    float* delta = aF(ph, PAIR_N);
    run_attn(ph, stream, d_in, pairT, zl, R, R, DP, 4,
             TAE_BASE + 0, TAE_BASE + 1, TAE_BASE + 2, TAE_BASE + 3, TAE_BASE + 4,
             TAE_BASE + 5, TAE_BASE + 6, TAE_BASE + 7, TAE_BASE + 8,
             PF(IN_TE_MASK), tb, nullptr, delta);
    add_transposed_kernel<<<EWG(PAIR_N), 0, stream>>>(pair, delta, R, R, DP);
  }

  // ---- 9) Pair transition -> pair ----
  run_transition(ws, stream, d_in, pair, (size_t)R * R, DP, 4 * DP, PT_BASE);
}
